// QRNN_1022202216451
// MI455X (gfx1250) — matrix operand order repaired and checked by reading, not compiled, not run on hardware
//
#include <hip/hip_runtime.h>

// Problem constants (from reference)
#define B_     32
#define T_     2048
#define D_     1024
#define H_     1024
#define OUT_   128
#define TCHUNK 128
#define NCHUNKS (T_ / TCHUNK)   // 16
#define NH     64               // hidden channels per workgroup (Z and F pair)
#define KSTEP  64               // K elements staged per LDS buffer

typedef __bf16 bf16_t;
typedef __attribute__((ext_vector_type(16))) __bf16 v16bf;
typedef __attribute__((ext_vector_type(8)))  __bf16 v8bf;
typedef __attribute__((ext_vector_type(8)))  float  v8f;

typedef unsigned int u32x4 __attribute__((ext_vector_type(4)));
typedef int          i32x8 __attribute__((ext_vector_type(8)));
typedef int          i32x4 __attribute__((ext_vector_type(4)));

__device__ __forceinline__ bf16_t f2bf(float f) {
  union { float f; unsigned u; } in; in.f = f;
  unsigned r = (in.u + 0x7fffu + ((in.u >> 16) & 1u)) >> 16;  // RNE
  unsigned short s = (unsigned short)r;
  bf16_t b; __builtin_memcpy(&b, &s, 2);
  return b;
}

// ---------------------------------------------------------------------------
// TDM: async 2D tile load Global -> LDS (tile_rows x tile_cols bf16 elements,
// source row stride row_stride_elems).  Descriptor per CDNA5 ISA ch.8.
// ---------------------------------------------------------------------------
__device__ __forceinline__ void tdm_load_2d(unsigned lds_off, const void* gaddr,
                                            unsigned tile_rows, unsigned tile_cols,
                                            unsigned row_stride_elems) {
  unsigned long long ga = (unsigned long long)(size_t)gaddr;
  u32x4 g0;
  g0[0] = 1u;                                   // count=1, user descriptor
  g0[1] = lds_off;                              // LDS byte address of tile start
  g0[2] = (unsigned)(ga & 0xFFFFFFFFu);         // global_addr[31:0]
  g0[3] = (unsigned)((ga >> 32) & 0x01FFFFFFu)  // global_addr[56:32]
          | 0x80000000u;                        // type=2 (image)
  i32x8 g1;
  g1[0] = (int)(1u << 16);                      // data_size=1 -> 2 bytes/elem
  g1[1] = (int)((row_stride_elems & 0xFFFFu) << 16);           // tensor_dim0[15:0]
  g1[2] = (int)((row_stride_elems >> 16) | (tile_rows << 16)); // dim0[31:16]|dim1[15:0]
  g1[3] = (int)(tile_cols << 16);               // tensor_dim1[31:16]=0 | tile_dim0
  g1[4] = (int)tile_rows;                       // tile_dim1 | tile_dim2=0
  g1[5] = (int)row_stride_elems;                // tensor_dim0_stride[31:0]
  g1[6] = 0;                                    // stride0[47:32] | stride1[15:0]
  g1[7] = 0;
  i32x4 z4 = (i32x4)0;
#if defined(__clang_major__) && (__clang_major__ >= 23)
  i32x8 z8 = (i32x8)0;
  __builtin_amdgcn_tensor_load_to_lds(g0, g1, z4, z4, z8, 0);
#else
  __builtin_amdgcn_tensor_load_to_lds(g0, g1, z4, z4, 0);
#endif
}

// ---------------------------------------------------------------------------
// Kernel 1: embedding gather + fp32 -> bf16 cast.  E[b,t,d] = emb[X[b,t], d]
// ---------------------------------------------------------------------------
__global__ __launch_bounds__(256)
void embed_cast_kernel(const int* __restrict__ X, const float* __restrict__ emb,
                       bf16_t* __restrict__ E) {
  long long gid = (long long)blockIdx.x * 256 + threadIdx.x;
  const long long total = (long long)B_ * T_ * (D_ / 8);
  if (gid >= total) return;
  int dp = (int)(gid % (D_ / 8));
  long long bt = gid / (D_ / 8);
  int tok = X[bt];
  const float4* src = (const float4*)(emb + (long long)tok * D_ + dp * 8);
  float4 a = src[0];
  float4 b = src[1];
  v8bf o;
  o[0] = f2bf(a.x); o[1] = f2bf(a.y); o[2] = f2bf(a.z); o[3] = f2bf(a.w);
  o[4] = f2bf(b.x); o[5] = f2bf(b.y); o[6] = f2bf(b.z); o[7] = f2bf(b.w);
  *(v8bf*)(E + bt * D_ + dp * 8) = o;
}

// ---------------------------------------------------------------------------
// Kernel 2: cast + repack Z/F rows of conv_w to bf16, blocked per channel
// group so each workgroup's B panel is one contiguous-row 2D tile:
//   Wpack[g][r][d],  r<64 -> Z row g*64+r,  r>=64 -> F row H + g*64 + (r-64)
// ---------------------------------------------------------------------------
__global__ __launch_bounds__(256)
void cast_w_kernel(const float* __restrict__ conv_w, bf16_t* __restrict__ Wpack) {
  long long gid = (long long)blockIdx.x * 256 + threadIdx.x;
  const long long total = (long long)2 * H_ * (D_ / 8);
  if (gid >= total) return;
  long long idx = gid * 8;
  int srow = (int)(idx / D_);          // 0..2H-1 (source row in conv_w)
  int scol = (int)(idx % D_);
  int drow;
  if (srow < H_) { int g = srow >> 6; int r = srow & 63; drow = g * 128 + r; }
  else { int s2 = srow - H_; int g = s2 >> 6; int r = s2 & 63; drow = g * 128 + 64 + r; }
  const float4* src = (const float4*)(conv_w + (long long)srow * D_ + scol);
  float4 a = src[0];
  float4 b = src[1];
  v8bf o;
  o[0] = f2bf(a.x); o[1] = f2bf(a.y); o[2] = f2bf(a.z); o[3] = f2bf(a.w);
  o[4] = f2bf(b.x); o[5] = f2bf(b.y); o[6] = f2bf(b.z); o[7] = f2bf(b.w);
  *(v8bf*)(Wpack + (long long)drow * D_ + scol) = o;
}

// ---------------------------------------------------------------------------
// Kernel 3: fused gate GEMM (WMMA bf16, TDM double-buffered 128x64 tiles) +
// bias + tanh/sigmoid + in-chunk fo-pool scan -> per-chunk affine map (P,C).
// Block: 256 threads (8 wave32).  Grid: (B*NCHUNKS, H/NH).
// ---------------------------------------------------------------------------
__global__ __launch_bounds__(256)
void gates_scan_kernel(const bf16_t* __restrict__ E,     // [B,T,D]
                       const bf16_t* __restrict__ Wpack, // [H/NH][128][D]
                       const float*  __restrict__ conv_b,
                       float* __restrict__ Pc,           // [NCHUNKS,B,H]
                       float* __restrict__ Cc)           // [NCHUNKS,B,H]
{
  const int bx   = blockIdx.x;
  const int b    = bx / NCHUNKS;
  const int ch   = bx % NCHUNKS;
  const int grp  = blockIdx.y;
  const int h0   = grp * NH;
  const int tid  = threadIdx.x;
  const int lane = tid & 31;
  const int wave = tid >> 5;            // 0..7, wave w owns rows [16w,16w+16)

  // GEMM staging buffers (64 KB) overlaid with scan staging (16.6 KB): the
  // scan phase begins only after the GEMM loop's final __syncthreads().
  __shared__ union SharedU {
    struct { bf16_t A[2][TCHUNK * KSTEP]; bf16_t B[2][128 * KSTEP]; } g;
    struct { float Z[32][NH + 1]; float F[32][NH + 1]; } s;
  } sh;

  v8f acc[8];
#pragma unroll
  for (int i = 0; i < 8; ++i) acc[i] = (v8f)0.0f;

  const bf16_t* Abase = E + ((long long)b * T_ + (long long)ch * TCHUNK) * D_;
  const bf16_t* Bbase = Wpack + (long long)grp * 128 * D_;

  // Prime TDM pipeline: tiles for k=0 into buffer 0.
  if (wave == 0) {
    tdm_load_2d((unsigned)(size_t)&sh.g.A[0][0], Abase, TCHUNK, KSTEP, D_);
    tdm_load_2d((unsigned)(size_t)&sh.g.B[0][0], Bbase, 128, KSTEP, D_);
  }

  int buf = 0;
  for (int k = 0; k < D_; k += KSTEP) {
    if (wave == 0) {
      if (k + KSTEP < D_) {
        // Prefetch next K-tiles into the other buffer, then wait only for the
        // current pair (tensor ops complete in order per wave).
        tdm_load_2d((unsigned)(size_t)&sh.g.A[buf ^ 1][0], Abase + (k + KSTEP), TCHUNK, KSTEP, D_);
        tdm_load_2d((unsigned)(size_t)&sh.g.B[buf ^ 1][0], Bbase + (k + KSTEP), 128, KSTEP, D_);
        __builtin_amdgcn_s_wait_tensorcnt(2);
      } else {
        __builtin_amdgcn_s_wait_tensorcnt(0);
      }
    }
    __syncthreads();

    const bf16_t* Abuf = &sh.g.A[buf][0];
    const bf16_t* Bbuf = &sh.g.B[buf][0];
    union fragu { v16bf v; v8bf h[2]; };
    const int m    = lane & 15;
    const int hsel = lane >> 4;
    const bf16_t* arow  = Abuf + (wave * 16 + m) * KSTEP;
    const bf16_t* blane = Bbuf + m * KSTEP + hsel * 8;

#pragma unroll
    for (int kh = 0; kh < 2; ++kh) {
      // A fragment (16-bit A 16x32 layout: lanes 0-15: M=lane, K={0..7,16..23};
      // lanes 16-31: M=lane-16, K={8..15,24..31}), K-half kh.
      fragu afrag;
      afrag.h[0] = *(const v8bf*)(arow + kh * 32 + hsel * 8);
      afrag.h[1] = *(const v8bf*)(arow + kh * 32 + 16 + hsel * 8);
      // Preload ALL 8 B fragments so ds_loads pipeline ahead of the WMMA chain.
      fragu bfr[8];
#pragma unroll
      for (int nt = 0; nt < 8; ++nt) {
        const bf16_t* bn = blane + nt * 16 * KSTEP + kh * 32;
        bfr[nt].h[0] = *(const v8bf*)(bn);
        bfr[nt].h[1] = *(const v8bf*)(bn + 16);
      }
#pragma unroll
      for (int nt = 0; nt < 8; ++nt) {
        acc[nt] = __builtin_amdgcn_wmma_f32_16x16x32_bf16(
            false, afrag.v, false, bfr[nt].v, (short)0, acc[nt], false, false);
      }
    }
    __syncthreads();
    buf ^= 1;
  }

  // Bias + activation + in-chunk scan.  C/D layout: VGPR j, lane<16 -> (M=j,
  // N=lane); lane>=16 -> (M=j+8, N=lane-16).  Stage 32 timesteps at a time.
  const int n    = lane & 15;
  const int msel = lane >> 4;
  float p = 1.0f, c = 0.0f;   // running chunk-affine map per scan thread

  for (int stage = 0; stage < TCHUNK / 32; ++stage) {
    if ((wave >> 1) == stage) {
#pragma unroll
      for (int nt = 0; nt < 8; ++nt) {
        int col  = nt * 16 + n;
        bool isF = col >= NH;
        int cc   = isF ? (col - NH) : col;
        float bias = conv_b[(isF ? H_ : 0) + h0 + cc];
#pragma unroll
        for (int j = 0; j < 8; ++j) {
          int mrow = (wave & 1) * 16 + j + msel * 8;     // row within stage
          float g = acc[nt][j] + bias;
          if (isF) sh.s.F[mrow][cc] = 1.0f / (1.0f + __expf(-g));
          else     sh.s.Z[mrow][cc] = tanhf(g);
        }
      }
    }
    __syncthreads();
    if (tid < NH) {
#pragma unroll 4
      for (int t = 0; t < 32; ++t) {
        float f = sh.s.F[t][tid];
        float z = sh.s.Z[t][tid];
        p = p * f;
        c = f * c + (1.0f - f) * z;
      }
    }
    __syncthreads();
  }

  if (tid < NH) {
    long long idx = ((long long)ch * B_ + b) * H_ + h0 + tid;
    Pc[idx] = p;
    Cc[idx] = c;
  }
}

// ---------------------------------------------------------------------------
// Kernel 4: sequential combine of chunk-affine maps:  h = P*h + C over chunks
// ---------------------------------------------------------------------------
__global__ __launch_bounds__(256)
void combine_kernel(const float* __restrict__ Pc, const float* __restrict__ Cc,
                    float* __restrict__ hfin) {
  int idx = blockIdx.x * 256 + threadIdx.x;     // over B*H
  if (idx >= B_ * H_) return;
  int b  = idx / H_;
  int hc = idx % H_;
  float h = 0.0f;
#pragma unroll
  for (int ch = 0; ch < NCHUNKS; ++ch) {
    long long o = ((long long)ch * B_ + b) * H_ + hc;
    h = Pc[o] * h + Cc[o];
  }
  hfin[idx] = h;
}

// ---------------------------------------------------------------------------
// Kernel 5: tiny output GEMM  out[b,o] = h[b,:] . out_w[o,:] + out_b[o]
// ---------------------------------------------------------------------------
__global__ __launch_bounds__(256)
void out_kernel(const float* __restrict__ hfin, const float* __restrict__ out_w,
                const float* __restrict__ out_b, float* __restrict__ out) {
  int idx = blockIdx.x * 256 + threadIdx.x;     // over B*OUT
  if (idx >= B_ * OUT_) return;
  int b = idx / OUT_;
  int o = idx % OUT_;
  const float* hv = hfin + (long long)b * H_;
  const float* wv = out_w + (long long)o * H_;
  float s = out_b[o];
#pragma unroll 8
  for (int k = 0; k < H_; ++k) s += hv[k] * wv[k];
  out[idx] = s;
}

// ---------------------------------------------------------------------------
extern "C" void kernel_launch(void* const* d_in, const int* in_sizes, int n_in,
                              void* d_out, int out_size, void* d_ws, size_t ws_size,
                              hipStream_t stream) {
  const int*   X      = (const int*)  d_in[0];   // [B,T] int32
  const float* emb    = (const float*)d_in[1];   // [V,D]
  const float* conv_w = (const float*)d_in[2];   // [3H,D]
  const float* conv_b = (const float*)d_in[3];   // [3H]
  const float* out_w  = (const float*)d_in[4];   // [OUT,H]
  const float* out_b  = (const float*)d_in[5];   // [OUT]
  float* out = (float*)d_out;                    // [B,OUT]

  // Workspace carve-up (~143 MB)
  char* p = (char*)d_ws;
  bf16_t* E   = (bf16_t*)p; p += (size_t)B_ * T_ * D_ * sizeof(bf16_t);
  bf16_t* Wpk = (bf16_t*)p; p += (size_t)2 * H_ * D_ * sizeof(bf16_t);
  float*  Pc  = (float*)p;  p += (size_t)NCHUNKS * B_ * H_ * sizeof(float);
  float*  Cc  = (float*)p;  p += (size_t)NCHUNKS * B_ * H_ * sizeof(float);
  float*  hf  = (float*)p;  p += (size_t)B_ * H_ * sizeof(float);

  {
    long long total = (long long)B_ * T_ * (D_ / 8);
    embed_cast_kernel<<<dim3((unsigned)((total + 255) / 256)), dim3(256), 0, stream>>>(X, emb, E);
  }
  {
    long long total = (long long)2 * H_ * (D_ / 8);
    cast_w_kernel<<<dim3((unsigned)((total + 255) / 256)), dim3(256), 0, stream>>>(conv_w, Wpk);
  }
  gates_scan_kernel<<<dim3(B_ * NCHUNKS, H_ / NH), dim3(256), 0, stream>>>(E, Wpk, conv_b, Pc, Cc);
  combine_kernel<<<dim3((B_ * H_ + 255) / 256), dim3(256), 0, stream>>>(Pc, Cc, hf);
  out_kernel<<<dim3((B_ * OUT_ + 255) / 256), dim3(256), 0, stream>>>(hf, out_w, out_b, out);
}
